// GraphNet_48000554500654
// MI455X (gfx1250) — matrix operand order repaired and checked by reading, not compiled, not run on hardware
//
#include <hip/hip_runtime.h>
#include <hip/hip_bf16.h>

typedef __attribute__((ext_vector_type(16))) _Float16 v16h;
typedef __attribute__((ext_vector_type(8)))  _Float16 v8h;
typedef __attribute__((ext_vector_type(4)))  _Float16 v4h;
typedef __attribute__((ext_vector_type(8)))  float    v8f;

// ---------------------------------------------------------------------------
// Zero a float buffer (grid-stride).
// ---------------------------------------------------------------------------
__global__ void zero_f32(float* __restrict__ p, size_t n) {
    size_t i = (size_t)blockIdx.x * blockDim.x + threadIdx.x;
    size_t stride = (size_t)gridDim.x * blockDim.x;
    for (; i < n; i += stride) p[i] = 0.0f;
}

// ---------------------------------------------------------------------------
// Pack a row-major f32 weight matrix W[K][Nc] into per-lane WMMA B-fragment
// layout (f16).  Fragment (nt, kt) covers B rows K = kt*32..kt*32+31 and
// columns N = nt*16..nt*16+15.  Contiguous k order used here, same as A (16-bit B 32x16): lane&15 = column,
// lanes 0-15 hold K = kt*32+0..15, lanes 16-31 hold K = kt*32+16..31,
// element i within the lane's 16 halves is consecutive K.
// out element index = ((nt*KT + kt)*32 + lane)*16 + i
// ---------------------------------------------------------------------------
__global__ void pack_wfrag(const float* __restrict__ W, int K, int Nc,
                           _Float16* __restrict__ out) {
    int idx = blockIdx.x * blockDim.x + threadIdx.x;
    if (idx >= K * Nc) return;
    int i    = idx & 15;
    int lane = (idx >> 4) & 31;
    int fk   = idx >> 9;            // = nt*KT + kt
    int KT   = K >> 5;
    int nt   = fk / KT;
    int kt   = fk - nt * KT;
    int n    = nt * 16 + (lane & 15);
    int k    = kt * 32 + ((lane >> 4) << 4) + i;
    out[idx] = (_Float16)W[(size_t)k * Nc + n];
}

// ---------------------------------------------------------------------------
// Edge scatter-add: agg[dst[e]][:] += x[src[e]][:]
// One thread per (edge, 4-feature chunk); float4 gather, 4 f32 atomics.
// Both agg accumulators (51/102 MB) fit in the 192 MB L2, so the atomics
// resolve in-cache.
// ---------------------------------------------------------------------------
template <int F>
__global__ void scatter_add(const float* __restrict__ x,
                            const int* __restrict__ src,
                            const int* __restrict__ dst,
                            float* __restrict__ agg, int nEdges) {
    constexpr int CH = F / 4;
    unsigned t = blockIdx.x * blockDim.x + threadIdx.x;
    unsigned total = (unsigned)nEdges * CH;
    if (t >= total) return;
    unsigned e = t / CH;
    unsigned c = (t - e * CH) * 4;
    int s = src[e];
    int d = dst[e];
    const float4 v = *(const float4*)(x + (size_t)s * F + c);
    float* o = agg + (size_t)d * F + c;
    atomicAdd(o + 0, v.x);
    atomicAdd(o + 1, v.y);
    atomicAdd(o + 2, v.z);
    atomicAdd(o + 3, v.w);
}

// ---------------------------------------------------------------------------
// Load one 16x32 f16 A-fragment from an LDS tile (row-major, stride S halves).
// Contiguous k order used here: lane<16 -> row=lane,    K = {0..15} of the 32-K slab
//             lane>=16 -> row=lane-16, K = {16..31}
// Two contiguous 16-byte LDS loads per lane.
// ---------------------------------------------------------------------------
__device__ inline v16h load_a_frag(const _Float16* buf, int S,
                                   int m, int hi, int kt) {
    const _Float16* p = buf + m * S + kt * 32 + hi * 16;
    v8h lo = *(const v8h*)p;
    v8h hs = *(const v8h*)(p + 8);
    return __builtin_shufflevector(lo, hs, 0, 1, 2, 3, 4, 5, 6, 7,
                                          8, 9, 10, 11, 12, 13, 14, 15);
}

// ---------------------------------------------------------------------------
// Fused GIN MLP:  out = relu( relu((x+agg) @ Wa + ba) @ Wb + bb )
// One wave (= one 32-thread block) computes NT=2 subtiles of 16 nodes each,
// so every B-fragment load from L2 feeds 2 WMMAs (M-direction reuse).
// A-fragments stream from per-block LDS slabs (ds_load_b128), where the
// WGP-local bandwidth has far more headroom than L2.
// ---------------------------------------------------------------------------
template <int FIN, int FOUT>
__global__ __launch_bounds__(32) void mlp_kernel(
    const float* __restrict__ xin, const float* __restrict__ agg,
    const v16h* __restrict__ WaF, const float* __restrict__ ba,
    const v16h* __restrict__ WbF, const float* __restrict__ bb,
    float* __restrict__ out, int nNodes) {
    constexpr int HIDc = 256;
    constexpr int NT   = 2;             // M-subtiles (16 rows each) per wave
    constexpr int SI   = FIN + 8;       // input row stride (halves), 16B pad
    constexpr int SH   = HIDc + 8;      // hidden row stride (halves)
    constexpr int KT1  = FIN / 32;
    constexpr int KT2  = HIDc / 32;

    __shared__ __align__(16) _Float16 xbuf[NT * 16 * SI];
    __shared__ __align__(16) _Float16 hbuf[NT * 16 * SH];

    const int lane = threadIdx.x;
    const int row0 = blockIdx.x * (NT * 16);
    if (row0 >= nNodes) return;

    // ---- stage (x + agg) rows into LDS as f16 (float4 loads, coalesced) ----
    for (int idx = lane; idx < NT * 16 * (FIN / 4); idx += 32) {
        int r  = idx / (FIN / 4);
        int c4 = idx - r * (FIN / 4);
        int gr = row0 + r;
        if (gr >= nNodes) gr = nNodes - 1;      // harmless duplicate on tail
        size_t g = (size_t)gr * (FIN / 4) + c4;
        const float4 a = ((const float4*)xin)[g];
        const float4 s = ((const float4*)agg)[g];
        v4h t;
        t[0] = (_Float16)(a.x + s.x);
        t[1] = (_Float16)(a.y + s.y);
        t[2] = (_Float16)(a.z + s.z);
        t[3] = (_Float16)(a.w + s.w);
        *(v4h*)(xbuf + r * SI + c4 * 4) = t;
    }
    __builtin_amdgcn_wave_barrier();

    const int m  = lane & 15;
    const int hi = lane >> 4;

    // ---- GEMM1: [NT*16 x FIN] @ [FIN x 256], bias + ReLU -> hbuf ----
    for (int nt = 0; nt < HIDc / 16; ++nt) {
        float bias = ba[nt * 16 + m];
        v8f acc[NT];
#pragma unroll
        for (int st = 0; st < NT; ++st)
#pragma unroll
            for (int j = 0; j < 8; ++j) acc[st][j] = bias;
#pragma unroll
        for (int kt = 0; kt < KT1; ++kt) {
            v16h bf = WaF[(nt * KT1 + kt) * 32 + lane];
#pragma unroll
            for (int st = 0; st < NT; ++st) {
                v16h a = load_a_frag(xbuf + st * 16 * SI, SI, m, hi, kt);
                acc[st] = __builtin_amdgcn_wmma_f32_16x16x32_f16(
                    false, a, false, bf, (short)0, acc[st], false, false);
            }
        }
        // C layout: lane column n = nt*16+m, VGPR j -> row j + hi*8
#pragma unroll
        for (int st = 0; st < NT; ++st)
#pragma unroll
            for (int j = 0; j < 8; ++j)
                hbuf[(st * 16 + j + hi * 8) * SH + nt * 16 + m] =
                    (_Float16)fmaxf(acc[st][j], 0.0f);
    }
    __builtin_amdgcn_wave_barrier();

    // ---- GEMM2: [NT*16 x 256] @ [256 x FOUT], bias + ReLU -> out ----
    for (int nt = 0; nt < FOUT / 16; ++nt) {
        float bias = bb[nt * 16 + m];
        v8f acc[NT];
#pragma unroll
        for (int st = 0; st < NT; ++st)
#pragma unroll
            for (int j = 0; j < 8; ++j) acc[st][j] = bias;
#pragma unroll
        for (int kt = 0; kt < KT2; ++kt) {
            v16h bf = WbF[(nt * KT2 + kt) * 32 + lane];
#pragma unroll
            for (int st = 0; st < NT; ++st) {
                v16h a = load_a_frag(hbuf + st * 16 * SH, SH, m, hi, kt);
                acc[st] = __builtin_amdgcn_wmma_f32_16x16x32_f16(
                    false, a, false, bf, (short)0, acc[st], false, false);
            }
        }
#pragma unroll
        for (int st = 0; st < NT; ++st)
#pragma unroll
            for (int j = 0; j < 8; ++j) {
                int gr = row0 + st * 16 + j + hi * 8;
                if (gr < nNodes)
                    out[(size_t)gr * FOUT + nt * 16 + m] =
                        fmaxf(acc[st][j], 0.0f);
            }
    }
}

// ---------------------------------------------------------------------------
// Orchestration
// ---------------------------------------------------------------------------
extern "C" void kernel_launch(void* const* d_in, const int* in_sizes, int n_in,
                              void* d_out, int out_size, void* d_ws, size_t ws_size,
                              hipStream_t stream) {
    const float* x   = (const float*)d_in[0];
    const int*   ei  = (const int*)d_in[1];
    const float* W1a = (const float*)d_in[2];
    const float* b1a = (const float*)d_in[3];
    const float* W1b = (const float*)d_in[4];
    const float* b1b = (const float*)d_in[5];
    const float* W2a = (const float*)d_in[6];
    const float* b2a = (const float*)d_in[7];
    const float* W2b = (const float*)d_in[8];
    const float* b2b = (const float*)d_in[9];

    const int N = in_sizes[0] / 128;
    const int E = in_sizes[1] / 2;
    const int* src = ei;
    const int* dst = ei + E;

    char* ws = (char*)d_ws;
    size_t off = 0;
    float* h1  = (float*)(ws + off); off += (size_t)N * 256 * sizeof(float);
    float* agg = (float*)(ws + off); off += (size_t)N * 256 * sizeof(float);
    v16h* w1aF = (v16h*)(ws + off);  off += (size_t)128 * 256 * sizeof(_Float16);
    v16h* w1bF = (v16h*)(ws + off);  off += (size_t)256 * 256 * sizeof(_Float16);
    v16h* w2aF = (v16h*)(ws + off);  off += (size_t)256 * 256 * sizeof(_Float16);
    v16h* w2bF = (v16h*)(ws + off);  off += (size_t)256 * 128 * sizeof(_Float16);

    // Weight fragment packing (tiny; L2-resident afterwards)
    pack_wfrag<<<(128 * 256 + 255) / 256, 256, 0, stream>>>(W1a, 128, 256, (_Float16*)w1aF);
    pack_wfrag<<<(256 * 256 + 255) / 256, 256, 0, stream>>>(W1b, 256, 256, (_Float16*)w1bF);
    pack_wfrag<<<(256 * 256 + 255) / 256, 256, 0, stream>>>(W2a, 256, 256, (_Float16*)w2aF);
    pack_wfrag<<<(256 * 128 + 255) / 256, 256, 0, stream>>>(W2b, 256, 128, (_Float16*)w2bF);

    const int grid = (N + 31) / 32;     // one 32-node wave-block per tile

    // ---- layer 1 ----
    zero_f32<<<2048, 256, 0, stream>>>(agg, (size_t)N * 128);
    {
        unsigned total = (unsigned)E * 32u;
        scatter_add<128><<<(total + 255) / 256, 256, 0, stream>>>(x, src, dst, agg, E);
    }
    mlp_kernel<128, 256><<<grid, 32, 0, stream>>>(x, agg, w1aF, b1a, w1bF, b1b, h1, N);

    // ---- layer 2 ----
    zero_f32<<<2048, 256, 0, stream>>>(agg, (size_t)N * 256);
    {
        unsigned total = (unsigned)E * 64u;
        scatter_add<256><<<(total + 255) / 256, 256, 0, stream>>>(h1, src, dst, agg, E);
    }
    mlp_kernel<256, 128><<<grid, 32, 0, stream>>>(h1, agg, w2aF, b2a, w2bF, b2b,
                                                  (float*)d_out, N);
}
